// BENDR_73942156968591
// MI455X (gfx1250) — matrix operand order repaired and checked by reading, not compiled, not run on hardware
//
#include <hip/hip_runtime.h>
#include <hip/hip_bf16.h>

#define B_ 16
#define F_ 256
#define T_ 2048
#define K_ 20

typedef __attribute__((ext_vector_type(16))) _Float16     v16h;
typedef __attribute__((ext_vector_type(8)))  _Float16     v8h;
typedef __attribute__((ext_vector_type(8)))  float        v8f;
typedef __attribute__((ext_vector_type(4)))  unsigned int v4u;

// -------------------------------------------------------------------------
// Kernel 1: transpose z[B,F,T] -> zh[B*T,F] (f16), c[b,:,t+1] -> ch[B*T,F],
// and compute exact fp32 row norms from the original fp32 data.
// One block handles (b, 16 consecutive t). Reads are 64B-contiguous along t,
// writes are fully coalesced 512B f16 rows.
// -------------------------------------------------------------------------
__global__ __launch_bounds__(256) void bendr_prep_kernel(
    const float* __restrict__ z, const float* __restrict__ c,
    _Float16* __restrict__ zh, _Float16* __restrict__ ch,
    float* __restrict__ znorm, float* __restrict__ cnorm)
{
    __shared__ float tile[16][F_ + 1];
    __shared__ float psum[16][16];

    const int blk = blockIdx.x;                // B * (T/16) blocks
    const int b   = blk / (T_ / 16);
    const int t0  = (blk % (T_ / 16)) * 16;
    const int tid = threadIdx.x;
    const int tc  = tid & 15;                  // t within tile
    const int fg  = tid >> 4;                  // f-group 0..15

    // ---------------- z ----------------
    float acc = 0.f;
    for (int r = 0; r < 16; ++r) {
        const int f = r * 16 + fg;
        const float v = z[((size_t)b * F_ + f) * T_ + t0 + tc];
        tile[tc][f] = v;
        acc += v * v;
    }
    psum[fg][tc] = acc;
    __syncthreads();
    if (tid < 16) {
        float s = 0.f;
        for (int g = 0; g < 16; ++g) s += psum[g][tid];
        znorm[(size_t)b * T_ + t0 + tid] = sqrtf(s);
    }
    for (int t = 0; t < 16; ++t)
        zh[((size_t)b * T_ + t0 + t) * F_ + tid] = (_Float16)tile[t][tid];
    __syncthreads();

    // ---------------- c (shifted by +1: drop start token) ----------------
    acc = 0.f;
    for (int r = 0; r < 16; ++r) {
        const int f = r * 16 + fg;
        const float v = c[((size_t)b * F_ + f) * (T_ + 1) + t0 + tc + 1];
        tile[tc][f] = v;
        acc += v * v;
    }
    psum[fg][tc] = acc;
    __syncthreads();
    if (tid < 16) {
        float s = 0.f;
        for (int g = 0; g < 16; ++g) s += psum[g][tid];
        cnorm[(size_t)b * T_ + t0 + tid] = sqrtf(s);
    }
    for (int t = 0; t < 16; ++t)
        ch[((size_t)b * T_ + t0 + t) * F_ + tid] = (_Float16)tile[t][tid];
}

// -------------------------------------------------------------------------
// Kernel 2: one wave32 per t. Targets (c_t + 20 gathered negatives) are the
// WMMA A matrix (2 M-tiles of 16 rows); z_t replicated across all 16 B
// columns. 8 k-steps of v_wmma_f32_16x16x32_f16 per tile. Equality mask is
// fused into the A-fragment loads (bit compare vs the c chunk).
// -------------------------------------------------------------------------
__global__ __launch_bounds__(256) void bendr_sim_kernel(
    const _Float16* __restrict__ zh, const _Float16* __restrict__ ch,
    const float* __restrict__ znorm, const float* __restrict__ cnorm,
    const int* __restrict__ negidx, float* __restrict__ out)
{
    __shared__ float colbuf[8][2][16];

    const int w    = threadIdx.x >> 5;         // wave in block (8 waves)
    const int lane = threadIdx.x & 31;
    const int m0   = lane & 15;                // row within M-tile
    const int h    = lane >> 4;                // lane half (K split)

    const int tIdx = blockIdx.x * 8 + w;       // 0 .. B*T-1
    const int b    = tIdx / T_;
    const int t    = tIdx % T_;

    const _Float16* zrow = zh + (size_t)tIdx * F_;
    const _Float16* crow = ch + (size_t)tIdx * F_;
    const size_t nbase   = (size_t)b * (T_ * K_) + (size_t)t * K_;

    // Per-lane target row pointers for the two M-tiles.
    // Tile0 rows = targets 0..15 (row 0 is the positive c_t).
    // Tile1 rows = targets 16..20; rows 21..31 padded with crow (discarded).
    const int  i0 = (m0 == 0) ? 0 : (m0 - 1);            // clamp: no OOB speculation
    const long r0 = negidx[nbase + i0];
    const _Float16* p0 = (m0 == 0) ? crow
                       : zh + ((size_t)b * T_ + (size_t)r0) * F_;
    const int  j1 = 16 + m0;
    const int  i1 = (j1 <= K_) ? (j1 - 1) : 0;
    const long r1 = negidx[nbase + i1];
    const _Float16* p1 = (j1 <= K_)
                       ? zh + ((size_t)b * T_ + (size_t)r1) * F_
                       : crow;

    v8f acc0 = {};
    v8f acc1 = {};
    int eq0 = 1, eq1 = 1;

    #pragma unroll
    for (int s = 0; s < 8; ++s) {
        const int f0 = s * 32;
        // B fragment: 32x16 f16, all 16 columns replicate z_t chunk.
        // Lane layout: elements 0..7 = B[h*8 + e][n], 8..15 = B[16 + h*8 + (e-8)][n].
        const v16h bz = __builtin_shufflevector(*(const v8h*)(zrow + f0 + h * 8), *(const v8h*)(zrow + f0 + 16 + h * 8), 0,1,2,3,4,5,6,7,8,9,10,11,12,13,14,15);

        // c chunk in A-fragment addressing (for the equality compare).
        const v8h c0 = *(const v8h*)(crow + f0 + h * 8);
        const v8h c1 = *(const v8h*)(crow + f0 + 16 + h * 8);

        // A fragments: per-lane two 16B half-chunks (K = h*8..h*8+7, 16+h*8..).
        const v8h a0 = *(const v8h*)(p0 + f0 + h * 8);
        const v8h a1 = *(const v8h*)(p0 + f0 + 16 + h * 8);
        const v8h d0 = *(const v8h*)(p1 + f0 + h * 8);
        const v8h d1 = *(const v8h*)(p1 + f0 + 16 + h * 8);

        // Fused exact-equality test (bitwise, zero extra memory traffic).
        {
            const v4u x = (__builtin_bit_cast(v4u, a0) ^ __builtin_bit_cast(v4u, c0))
                        | (__builtin_bit_cast(v4u, a1) ^ __builtin_bit_cast(v4u, c1));
            eq0 &= ((x.x | x.y | x.z | x.w) == 0u);
            const v4u y = (__builtin_bit_cast(v4u, d0) ^ __builtin_bit_cast(v4u, c0))
                        | (__builtin_bit_cast(v4u, d1) ^ __builtin_bit_cast(v4u, c1));
            eq1 &= ((y.x | y.y | y.z | y.w) == 0u);
        }

        const v16h A0 = __builtin_shufflevector(a0, a1, 0,1,2,3,4,5,6,7,8,9,10,11,12,13,14,15);
        const v16h A1 = __builtin_shufflevector(d0, d1, 0,1,2,3,4,5,6,7,8,9,10,11,12,13,14,15);

        acc0 = __builtin_amdgcn_wmma_f32_16x16x32_f16(false, A0, false, bz,
                                                      (short)0, acc0, false, false);
        acc1 = __builtin_amdgcn_wmma_f32_16x16x32_f16(false, A1, false, bz,
                                                      (short)0, acc1, false, false);
    }

    // Row equality = AND of both K-halves of the row (lanes m0 and m0+16).
    eq0 &= __shfl_xor(eq0, 16, 32);
    eq1 &= __shfl_xor(eq1, 16, 32);
    const unsigned bal0 = (unsigned)__ballot(eq0);  // bits 0..15 = tile0 rows
    const unsigned bal1 = (unsigned)__ballot(eq1);  // bits 0..15 = tile1 rows

    // All B columns were identical, so D[m][n] is the same for every n;
    // lanes 0 (rows 0..7) and 16 (rows 8..15) publish the 16 row dots.
    if (m0 == 0) {
        #pragma unroll
        for (int v = 0; v < 8; ++v) {
            colbuf[w][0][h * 8 + v] = acc0[v];
            colbuf[w][1][h * 8 + v] = acc1[v];
        }
    }
    __syncthreads();

    if (lane <= K_) {
        const int j   = lane;                    // target 0..20
        const float num = (j < 16) ? colbuf[w][0][j] : colbuf[w][1][j - 16];
        const float zn  = znorm[tIdx];
        const int  ij   = (j == 0) ? 0 : (j - 1);
        const long ridx = negidx[nbase + ij];
        const float tn  = (j == 0) ? cnorm[tIdx]
                        : znorm[(size_t)b * T_ + (size_t)ridx];
        const float denom = fmaxf(zn * tn, 1e-8f);
        const float logit = (num / denom) * 2.0f;          // / TEMP (0.5)
        const bool masked = (j > 0) &&
            ((((j < 16) ? (bal0 >> j) : (bal1 >> (j - 16))) & 1u) != 0u);
        out[(size_t)tIdx * (K_ + 1) + j] = masked ? -__builtin_huge_valf() : logit;
    }
}

// -------------------------------------------------------------------------
extern "C" void kernel_launch(void* const* d_in, const int* in_sizes, int n_in,
                              void* d_out, int out_size, void* d_ws, size_t ws_size,
                              hipStream_t stream) {
    const float* z      = (const float*)d_in[0];   // [B, F, T]
    const float* c      = (const float*)d_in[1];   // [B, F, T+1]
    const int*   negidx = (const int*)d_in[2];     // [B, T*K]
    float*       out    = (float*)d_out;           // [B*T, K+1]

    char* ws = (char*)d_ws;
    const size_t ZH = (size_t)B_ * T_ * F_ * sizeof(_Float16);   // 16 MB
    const size_t NB = (size_t)B_ * T_ * sizeof(float);           // 128 KB
    _Float16* zh    = (_Float16*)(ws);
    _Float16* ch    = (_Float16*)(ws + ZH);
    float*    znorm = (float*)(ws + 2 * ZH);
    float*    cnorm = (float*)(ws + 2 * ZH + NB);

    bendr_prep_kernel<<<B_ * (T_ / 16), 256, 0, stream>>>(z, c, zh, ch, znorm, cnorm);
    bendr_sim_kernel<<<(B_ * T_) / 8, 256, 0, stream>>>(zh, ch, znorm, cnorm, negidx, out);
}
